// ContraNorm_55405078118436
// MI455X (gfx1250) — matrix operand order repaired and checked by reading, not compiled, not run on hardware
//
#include <hip/hip_runtime.h>
#include <hip/hip_bf16.h>

// ---------------------------------------------------------------------------
// ContraNorm, MI455X (gfx1250), wave32.
//   - v_wmma_f32_16x16x32_bf16 with error-compensated hi/lo bf16 split
//     (hi*hi + hi*lo + lo*hi  ~= fp32 accuracy at bf16 matrix rates)
//   - a[n,m] = xn.xn/TEMP is symmetric and bounded by 2 -> constant-shift
//     softmax (no online max), column pass folded into a row pass.
//   - 4 waves / workgroup share each column tile, staged in LDS through the
//     CDNA5 async-load path (global_load_async_to_lds_b128 + s_wait_asynccnt)
//     and double-buffered against WMMA compute: cuts L2 panel traffic 4x,
//     which is what moves this kernel from L2-BW-bound to matrix-bound.
// ---------------------------------------------------------------------------

typedef __bf16 bf16;
typedef __attribute__((ext_vector_type(16))) __bf16 v16bf;
typedef __attribute__((ext_vector_type(8)))  __bf16 v8bf;
typedef __attribute__((ext_vector_type(8)))  float  v8f;

static constexpr int B = 4;
static constexpr int N = 4096;
static constexpr int D = 128;
static constexpr float INV_TEMP = 2.0f;   // 1/0.5
static constexpr float SCALE    = 0.1f;
static constexpr float LN_EPS   = 1e-6f;

#define WMMA_BF16(A_, B_, C_) \
  __builtin_amdgcn_wmma_f32_16x16x32_bf16(false, (A_), false, (B_), (short)0, (C_), false, false)

// ---- CDNA5 async global->LDS (ASYNCcnt) -----------------------------------
// Builtin signature (from hipcc diagnostic): (int4 AS1*, int4 AS3*, Ii, Ii)
typedef int v4i __attribute__((vector_size(16)));
typedef __attribute__((address_space(1))) v4i gv4i;
typedef __attribute__((address_space(3))) v4i lv4i;

#if __has_builtin(__builtin_amdgcn_global_load_async_to_lds_b128)
__device__ __forceinline__ void async_b128(const void* g, void* l) {
  __builtin_amdgcn_global_load_async_to_lds_b128((gv4i*)g, (lv4i*)l, 0, 0);
}
#else
__device__ __forceinline__ void async_b128(const void* g, void* l) {
  unsigned lds = (unsigned)(unsigned long long)l;   // LDS byte offset = low bits
  asm volatile("global_load_async_to_lds_b128 %0, %1, off"
               :: "v"(lds), "v"(g) : "memory");
}
#endif

template <int Nw>
__device__ __forceinline__ void wait_async() {
#if __has_builtin(__builtin_amdgcn_s_wait_asynccnt)
  __builtin_amdgcn_s_wait_asynccnt(Nw);
#else
  asm volatile("s_wait_asynccnt %0" :: "i"(Nw) : "memory");
#endif
}

// Load 16 bf16 as two contiguous 16B chunks -> v16bf fragment (works for
// global or LDS pointers; LDS side lowers to ds_load_b128).
__device__ __forceinline__ v16bf ld16(const bf16* p0, const bf16* p1) {
  v8bf a = *(const v8bf*)p0;
  v8bf b = *(const v8bf*)p1;
  v16bf r;
#pragma unroll
  for (int e = 0; e < 8; ++e) { r[e] = a[e]; r[e + 8] = b[e]; }
  return r;
}

// A fragment (16x32 bf16): lane holds row l15, contiguous K
//   [kc*32 + half*16 .. +15].
__device__ __forceinline__ void load_afrags(const bf16* Ah, const bf16* Al,
                                            int half, v16bf* Ahi, v16bf* Alo) {
#pragma unroll
  for (int kc = 0; kc < 4; ++kc) {
    Ahi[kc] = ld16(Ah + kc * 32 + half * 16, Ah + kc * 32 + half * 16 + 8);
    Alo[kc] = ld16(Al + kc * 32 + half * 16, Al + kc * 32 + half * 16 + 8);
  }
}

// One 16x16 sim tile vs a B row (col) of 128 K values (hi/lo split).
// B fragment (32x16 bf16): lane = col l15, contiguous K [kc*32+half*16 .. +15].
__device__ __forceinline__ v8f sim_tile(const v16bf* Ahi, const v16bf* Alo,
                                        const bf16* Bh_row, const bf16* Bl_row,
                                        int half) {
  v8f acc0 = {0.f, 0.f, 0.f, 0.f, 0.f, 0.f, 0.f, 0.f};
  v8f acc1 = {0.f, 0.f, 0.f, 0.f, 0.f, 0.f, 0.f, 0.f};
#pragma unroll
  for (int kc = 0; kc < 4; ++kc) {
    const bf16* ph = Bh_row + kc * 32 + half * 16;
    const bf16* pl = Bl_row + kc * 32 + half * 16;
    v16bf bh = ld16(ph, ph + 8);
    v16bf bl = ld16(pl, pl + 8);
    acc0 = WMMA_BF16(Ahi[kc], bh, acc0);   // hi*hi
    acc1 = WMMA_BF16(Ahi[kc], bl, acc1);   // hi*lo
    acc1 = WMMA_BF16(Alo[kc], bh, acc1);   // lo*hi
  }
#pragma unroll
  for (int v = 0; v < 8; ++v) acc0[v] += acc1[v];
  return acc0;
}

// ---- kernel 1: L2-normalize rows + bf16 hi/lo pack (row-major) ------------
__global__ void __launch_bounds__(32)
cn_prep(const float* __restrict__ x,
        bf16* __restrict__ xn_hi, bf16* __restrict__ xn_lo) {
  const int row = blockIdx.x;              // flat [b][n]
  const int lane = threadIdx.x;
  const float* xr = x + (size_t)row * D;
  float v[4], ss = 0.f;
#pragma unroll
  for (int i = 0; i < 4; ++i) { v[i] = xr[lane + 32 * i]; ss += v[i] * v[i]; }
#pragma unroll
  for (int off = 16; off; off >>= 1) ss += __shfl_xor(ss, off, 32);
  const float rn = 1.f / fmaxf(sqrtf(ss), 1e-12f);
#pragma unroll
  for (int i = 0; i < 4; ++i) {
    const int d = lane + 32 * i;
    const float xnv = v[i] * rn;
    bf16 h = (bf16)xnv;
    xn_hi[(size_t)row * D + d] = h;
    xn_lo[(size_t)row * D + d] = (bf16)(xnv - (float)h);
  }
}

// ---- kernel 2: bf16 hi/lo pack of raw x, transposed to [b][d][n] ----------
__global__ void __launch_bounds__(256)
cn_transpose(const float* __restrict__ x,
             bf16* __restrict__ xt_hi, bf16* __restrict__ xt_lo) {
  __shared__ float tile[32][33];
  const int bid = blockIdx.x;              // [b][nTile(128)][dTile(4)]
  const int dT = bid & 3;
  const int nT = (bid >> 2) & 127;
  const int b  = bid >> 9;
  const int tx = threadIdx.x & 31;
  const int ty = threadIdx.x >> 5;         // 0..7
  const int n0 = nT * 32, d0 = dT * 32;
#pragma unroll
  for (int i = 0; i < 4; ++i)
    tile[ty + 8 * i][tx] = x[((size_t)b * N + n0 + ty + 8 * i) * D + d0 + tx];
  __syncthreads();
#pragma unroll
  for (int i = 0; i < 4; ++i) {
    const int d = d0 + ty + 8 * i;
    const float v = tile[tx][ty + 8 * i];
    bf16 h = (bf16)v;
    xt_hi[((size_t)b * D + d) * N + n0 + tx] = h;
    xt_lo[((size_t)b * D + d) * N + n0 + tx] = (bf16)(v - (float)h);
  }
}

// ---- shared helper: async-copy one contiguous 16x128 bf16 panel (4KB) -----
__device__ __forceinline__ void issue_tile16(const bf16* gh, const bf16* gl,
                                             bf16* lh, bf16* ll, int tid) {
  const char* gh8 = (const char*)gh; const char* gl8 = (const char*)gl;
  char* lh8 = (char*)lh;             char* ll8 = (char*)ll;
  async_b128(gh8 + tid * 32,      lh8 + tid * 32);
  async_b128(gh8 + tid * 32 + 16, lh8 + tid * 32 + 16);
  async_b128(gl8 + tid * 32,      ll8 + tid * 32);
  async_b128(gl8 + tid * 32 + 16, ll8 + tid * 32 + 16);
}

// ---- kernel 3: S_n = sum_m exp(a[n,m] - 2) --------------------------------
__global__ void __launch_bounds__(128)
cn_pass1(const bf16* __restrict__ xn_hi, const bf16* __restrict__ xn_lo,
         float* __restrict__ Ssum) {
  __shared__ __align__(16) bf16 sB[2][2][16 * D];   // 16KB, double-buffered
  const int blk = blockIdx.x;
  const int b   = blk / (N / 64);
  const int tid = threadIdx.x;
  const int rb  = (blk % (N / 64)) * 64 + (tid >> 5) * 16;
  const int lane = tid & 31, half = lane >> 4, l15 = lane & 15;

  v16bf Ahi[4], Alo[4];
  load_afrags(xn_hi + ((size_t)b * N + rb + l15) * D,
              xn_lo + ((size_t)b * N + rb + l15) * D, half, Ahi, Alo);

  issue_tile16(xn_hi + (size_t)b * N * D, xn_lo + (size_t)b * N * D,
               sB[0][0], sB[0][1], tid);

  float s[8];
#pragma unroll
  for (int v = 0; v < 8; ++v) s[v] = 0.f;

  int p = 0;
  for (int j = 0; j < N; j += 16) {
    wait_async<0>();
    __syncthreads();                       // tile j visible; prior reads done
    if (j + 16 < N)
      issue_tile16(xn_hi + ((size_t)b * N + j + 16) * D,
                   xn_lo + ((size_t)b * N + j + 16) * D,
                   sB[p ^ 1][0], sB[p ^ 1][1], tid);
    v8f acc = sim_tile(Ahi, Alo, sB[p][0] + l15 * D, sB[p][1] + l15 * D, half);
#pragma unroll
    for (int v = 0; v < 8; ++v) s[v] += __expf(fmaf(acc[v], INV_TEMP, -2.f));
    p ^= 1;
  }
#pragma unroll
  for (int v = 0; v < 8; ++v) {
    float sv = s[v];
#pragma unroll
    for (int off = 8; off; off >>= 1) sv += __shfl_xor(sv, off, 32);
    if (l15 == 0) Ssum[(size_t)b * N + rb + v + 8 * half] = sv;
  }
}

// ---- kernel 4: C_m = sum_n exp(exp(a[m,n]-2)/S_n)  (a symmetric) ----------
__global__ void __launch_bounds__(128)
cn_pass2(const bf16* __restrict__ xn_hi, const bf16* __restrict__ xn_lo,
         const float* __restrict__ Ssum, float* __restrict__ Csum) {
  __shared__ __align__(16) bf16 sB[2][2][16 * D];
  const int blk = blockIdx.x;
  const int b   = blk / (N / 64);
  const int tid = threadIdx.x;
  const int rb  = (blk % (N / 64)) * 64 + (tid >> 5) * 16;
  const int lane = tid & 31, half = lane >> 4, l15 = lane & 15;

  v16bf Ahi[4], Alo[4];
  load_afrags(xn_hi + ((size_t)b * N + rb + l15) * D,
              xn_lo + ((size_t)b * N + rb + l15) * D, half, Ahi, Alo);

  issue_tile16(xn_hi + (size_t)b * N * D, xn_lo + (size_t)b * N * D,
               sB[0][0], sB[0][1], tid);

  float c[8];
#pragma unroll
  for (int v = 0; v < 8; ++v) c[v] = 0.f;

  int p = 0;
  for (int j = 0; j < N; j += 16) {
    wait_async<0>();
    __syncthreads();
    if (j + 16 < N)
      issue_tile16(xn_hi + ((size_t)b * N + j + 16) * D,
                   xn_lo + ((size_t)b * N + j + 16) * D,
                   sB[p ^ 1][0], sB[p ^ 1][1], tid);
    v8f acc = sim_tile(Ahi, Alo, sB[p][0] + l15 * D, sB[p][1] + l15 * D, half);
    const float rS = 1.f / Ssum[(size_t)b * N + j + l15];  // per-column 1/S
#pragma unroll
    for (int v = 0; v < 8; ++v)
      c[v] += __expf(__expf(fmaf(acc[v], INV_TEMP, -2.f)) * rS);
    p ^= 1;
  }
#pragma unroll
  for (int v = 0; v < 8; ++v) {
    float cv = c[v];
#pragma unroll
    for (int off = 8; off; off >>= 1) cv += __shfl_xor(cv, off, 32);
    if (l15 == 0) Csum[(size_t)b * N + rb + v + 8 * half] = cv;
  }
}

// ---- kernel 5: x_neg = (sim + colsoftmax(sim)) @ x ------------------------
__global__ void __launch_bounds__(128)
cn_pass3(const bf16* __restrict__ xn_hi, const bf16* __restrict__ xn_lo,
         const bf16* __restrict__ xt_hi, const bf16* __restrict__ xt_lo,
         const float* __restrict__ Ssum, const float* __restrict__ Csum,
         float* __restrict__ xneg) {
  __shared__ __align__(16) bf16 sB[2][2][32 * D];   // sim B panel, 32KB (2 bufs)
  __shared__ __align__(16) bf16 sX[2][D * 32];      // x^T panel [d][32], 16KB
  __shared__ __align__(16) bf16 sW[2][4][16 * 32];  // per-wave W tile, 8KB
  const int blk = blockIdx.x;
  const int b   = blk / (N / 64);
  const int tid = threadIdx.x;
  const int w   = tid >> 5;
  const int rb  = (blk % (N / 64)) * 64 + w * 16;
  const int lane = tid & 31, half = lane >> 4, l15 = lane & 15;

  v16bf Ahi[4], Alo[4];
  load_afrags(xn_hi + ((size_t)b * N + rb + l15) * D,
              xn_lo + ((size_t)b * N + rb + l15) * D, half, Ahi, Alo);

  float rS[8];
#pragma unroll
  for (int v = 0; v < 8; ++v)
    rS[v] = 1.f / Ssum[(size_t)b * N + rb + v + 8 * half];

  v8f out[8];
#pragma unroll
  for (int t = 0; t < 8; ++t)
#pragma unroll
    for (int v = 0; v < 8; ++v) out[t][v] = 0.f;

  // preload sim panel j=0: 8KB contiguous per hi/lo -> 64B per thread
  {
    const char* gh = (const char*)(xn_hi + (size_t)b * N * D);
    const char* gl = (const char*)(xn_lo + (size_t)b * N * D);
#pragma unroll
    for (int k = 0; k < 4; ++k) {
      async_b128(gh + tid * 64 + k * 16, (char*)sB[0][0] + tid * 64 + k * 16);
      async_b128(gl + tid * 64 + k * 16, (char*)sB[0][1] + tid * 64 + k * 16);
    }
  }

  int p = 0;
  for (int j = 0; j < N; j += 32) {
    wait_async<0>();
    __syncthreads();                 // sim panel j visible; prior sX reads done

    // issue x^T panel for current j: thread tid copies d-row tid (64B/array)
    {
      const char* gh = (const char*)(xt_hi + ((size_t)b * D + tid) * N + j);
      const char* gl = (const char*)(xt_lo + ((size_t)b * D + tid) * N + j);
#pragma unroll
      for (int k = 0; k < 4; ++k) {
        async_b128(gh + k * 16, (char*)sX[0] + tid * 64 + k * 16);
        async_b128(gl + k * 16, (char*)sX[1] + tid * 64 + k * 16);
      }
    }
    const bool more = (j + 32 < N);
    if (more) {                      // prefetch next sim panel (other buffer)
      const char* gh = (const char*)(xn_hi + ((size_t)b * N + j + 32) * D);
      const char* gl = (const char*)(xn_lo + ((size_t)b * N + j + 32) * D);
#pragma unroll
      for (int k = 0; k < 4; ++k) {
        async_b128(gh + tid * 64 + k * 16, (char*)sB[p ^ 1][0] + tid * 64 + k * 16);
        async_b128(gl + tid * 64 + k * 16, (char*)sB[p ^ 1][1] + tid * 64 + k * 16);
      }
    }

    // two 16x16 sim tiles from LDS panel (overlaps the x^T async load)
    v8f a0 = sim_tile(Ahi, Alo, sB[p][0] + l15 * D,        sB[p][1] + l15 * D,        half);
    v8f a1 = sim_tile(Ahi, Alo, sB[p][0] + (16 + l15) * D, sB[p][1] + (16 + l15) * D, half);
    const float rC0 = 1.f / Csum[(size_t)b * N + j + l15];
    const float rC1 = 1.f / Csum[(size_t)b * N + j + 16 + l15];

    // W = sim + exp(sim)/C, hi/lo split, staged per-wave in LDS
#pragma unroll
    for (int v = 0; v < 8; ++v) {
      const int r = v + 8 * half;
      float e0 = __expf(fmaf(a0[v], INV_TEMP, -2.f)) * rS[v];
      float w0 = e0 + __expf(e0) * rC0;
      float e1 = __expf(fmaf(a1[v], INV_TEMP, -2.f)) * rS[v];
      float w1 = e1 + __expf(e1) * rC1;
      bf16 h0 = (bf16)w0;
      sW[0][w][r * 32 + l15] = h0;       sW[1][w][r * 32 + l15] = (bf16)(w0 - (float)h0);
      bf16 h1 = (bf16)w1;
      sW[0][w][r * 32 + 16 + l15] = h1;  sW[1][w][r * 32 + 16 + l15] = (bf16)(w1 - (float)h1);
    }

    // async loads complete in order: waiting <=8 retires the 8 x^T ops while
    // the 8 prefetched sim-panel ops stay in flight.
    if (more) wait_async<8>(); else wait_async<0>();
    __syncthreads();                 // x^T panel + W tiles visible

    const bf16* wbh = sW[0][w] + l15 * 32;
    const bf16* wbl = sW[1][w] + l15 * 32;
    v16bf wh = ld16(wbh + half * 16, wbh + half * 16 + 8);
    v16bf wl = ld16(wbl + half * 16, wbl + half * 16 + 8);

    // out[16 x 128] += W(16x32) @ X(32x128), X fragments from LDS
#pragma unroll
    for (int t = 0; t < 8; ++t) {
      const bf16* Xh = sX[0] + (t * 16 + l15) * 32 + half * 16;
      const bf16* Xl = sX[1] + (t * 16 + l15) * 32 + half * 16;
      v16bf bh = ld16(Xh, Xh + 8);
      v16bf bl = ld16(Xl, Xl + 8);
      out[t] = WMMA_BF16(wh, bh, out[t]);
      out[t] = WMMA_BF16(wh, bl, out[t]);
      out[t] = WMMA_BF16(wl, bh, out[t]);
    }
    p ^= 1;
  }

#pragma unroll
  for (int t = 0; t < 8; ++t)
#pragma unroll
    for (int v = 0; v < 8; ++v)
      xneg[((size_t)b * N + rb + v + 8 * half) * D + t * 16 + l15] = out[t][v];
}

// ---- kernel 6: y = x - SCALE*x_neg; LayerNorm(d=128, eps=1e-6) ------------
__global__ void __launch_bounds__(32)
cn_ln(const float* __restrict__ x, const float* __restrict__ xneg,
      const float* __restrict__ w, const float* __restrict__ bias,
      float* __restrict__ out) {
  const int row = blockIdx.x;
  const int lane = threadIdx.x;
  const float* xr = x + (size_t)row * D;
  const float* nr = xneg + (size_t)row * D;
  float y[4], mu = 0.f;
#pragma unroll
  for (int i = 0; i < 4; ++i) {
    const int d = lane + 32 * i;
    y[i] = xr[d] - SCALE * nr[d];
    mu += y[i];
  }
#pragma unroll
  for (int off = 16; off; off >>= 1) mu += __shfl_xor(mu, off, 32);
  mu *= (1.f / D);
  float var = 0.f;
#pragma unroll
  for (int i = 0; i < 4; ++i) { const float t = y[i] - mu; var += t * t; }
#pragma unroll
  for (int off = 16; off; off >>= 1) var += __shfl_xor(var, off, 32);
  var *= (1.f / D);
  const float rs = rsqrtf(var + LN_EPS);
#pragma unroll
  for (int i = 0; i < 4; ++i) {
    const int d = lane + 32 * i;
    out[(size_t)row * D + d] = (y[i] - mu) * rs * w[d] + bias[d];
  }
}

// ---------------------------------------------------------------------------
extern "C" void kernel_launch(void* const* d_in, const int* in_sizes, int n_in,
                              void* d_out, int out_size, void* d_ws, size_t ws_size,
                              hipStream_t stream) {
  (void)in_sizes; (void)n_in; (void)out_size; (void)ws_size;
  const float* x   = (const float*)d_in[0];
  const float* lnw = (const float*)d_in[1];
  const float* lnb = (const float*)d_in[2];
  float* out = (float*)d_out;

  char* ws = (char*)d_ws;
  size_t off = 0;
  bf16* xn_hi = (bf16*)(ws + off); off += (size_t)B * N * D * sizeof(bf16);
  bf16* xn_lo = (bf16*)(ws + off); off += (size_t)B * N * D * sizeof(bf16);
  bf16* xt_hi = (bf16*)(ws + off); off += (size_t)B * N * D * sizeof(bf16);
  bf16* xt_lo = (bf16*)(ws + off); off += (size_t)B * N * D * sizeof(bf16);
  float* Ssum = (float*)(ws + off); off += (size_t)B * N * sizeof(float);
  float* Csum = (float*)(ws + off); off += (size_t)B * N * sizeof(float);
  float* xneg = (float*)(ws + off); off += (size_t)B * N * D * sizeof(float);

  cn_prep      <<<B * N,            32, 0, stream>>>(x, xn_hi, xn_lo);
  cn_transpose <<<B * (N/32)*(D/32), 256, 0, stream>>>(x, xt_hi, xt_lo);
  cn_pass1     <<<B * (N / 64),     128, 0, stream>>>(xn_hi, xn_lo, Ssum);
  cn_pass2     <<<B * (N / 64),     128, 0, stream>>>(xn_hi, xn_lo, Ssum, Csum);
  cn_pass3     <<<B * (N / 64),     128, 0, stream>>>(xn_hi, xn_lo, xt_hi, xt_lo,
                                                      Ssum, Csum, xneg);
  cn_ln        <<<B * N,            32, 0, stream>>>(x, xneg, lnw, lnb, out);
}
